// HierarchicalGNN_73100343378657
// MI455X (gfx1250) — matrix operand order repaired and checked by reading, not compiled, not run on hardware
//
#include <hip/hip_runtime.h>
#include <hip/hip_bf16.h>
#include <math.h>

#define N_NODES 16000
#define N_EDGES 256000
#define EPRIME  (N_EDGES + N_NODES)   // GATv2 adds self-loops
#define DIN     1280
#define HID     256
#define HEADS   8
#define CH      32
#define NGRAPH  32
#define NLAB    64
#define EPSF    1e-5f

typedef __attribute__((ext_vector_type(16))) _Float16 v16h;
typedef __attribute__((ext_vector_type(8)))  float    v8f;

// ---------- helpers ----------
__device__ __forceinline__ unsigned fkey(float f) {
  unsigned u = __float_as_uint(f);
  return (u & 0x80000000u) ? ~u : (u | 0x80000000u);
}
__device__ __forceinline__ float funkey(unsigned u) {
  return (u & 0x80000000u) ? __uint_as_float(u & 0x7FFFFFFFu) : __uint_as_float(~u);
}
__device__ __forceinline__ float leaky(float v) { return v > 0.f ? v : 0.2f * v; }

// ---------- fills ----------
__global__ void fill_f32(float* p, float v, int n) {
  int i = blockIdx.x * blockDim.x + threadIdx.x;
  if (i < n) p[i] = v;
}
__global__ void fill_key(unsigned* p, float v, int n) {
  int i = blockIdx.x * blockDim.x + threadIdx.x;
  if (i < n) p[i] = fkey(v);
}

// ---------- WMMA GEMM: out = act(A[MxK] @ W[KxN] + bias) ----------
enum { ACT_NONE = 0, ACT_RELU_BN = 1, ACT_RELU_BN_LEAKY = 2, ACT_TANH = 3, ACT_SILU_BN = 4 };

#define BROW 18   // padded LDS row stride (floats): even (8B-aligned b64 dests),
                  // and 8*18=144 % 64 = 16 -> no bank conflict between lane L and L+16

template <int MODE>
__global__ __launch_bounds__(256) void wmma_gemm(
    const float* __restrict__ A, const float* __restrict__ W,
    const float* __restrict__ bias, const float* __restrict__ gg,
    const float* __restrict__ bb, float* __restrict__ out,
    int M, int K, int Nout,
    long sA, long sW, long sBias, long sAff, long sOut) {
  __shared__ float Bsh[32 * BROW];   // 32x16 fp32 W tile, padded rows
  int z = blockIdx.z;
  A   += (size_t)z * sA;
  W   += (size_t)z * sW;
  out += (size_t)z * sOut;
  const float* biasZ = bias ? bias + (size_t)z * sBias : nullptr;
  const float* ggZ = gg, *bbZ = bb;
  if (MODE == ACT_RELU_BN || MODE == ACT_RELU_BN_LEAKY || MODE == ACT_SILU_BN) {
    ggZ = gg + (size_t)z * sAff;
    bbZ = bb + (size_t)z * sAff;
  }

  int lane = threadIdx.x & 31;
  int wave = threadIdx.x >> 5;
  int half = lane >> 4;   // 0 or 1
  int l16  = lane & 15;
  int m0 = (blockIdx.y * 8 + wave) * 16;
  int n0 = blockIdx.x * 16;
  bool active = (m0 < M);   // wave-uniform (M is a multiple of 16)

  // per-thread async-copy assignment: thread t copies floats (kk, nn) and (kk, nn+1)
  int pr  = 2 * threadIdx.x;          // 0..510, even
  int skk = pr >> 4;                  // 0..31
  int snn = pr & 15;                  // even
  unsigned lds_off = (unsigned)(size_t)(&Bsh[skk * BROW + snn]);  // LDS byte offset = addr[31:0]

  v8f acc = {};
  for (int k0 = 0; k0 < K; k0 += 32) {
    __syncthreads();   // previous tile fully consumed before overwrite
    {
      // async copy memory -> LDS (8 bytes per thread; tracked by ASYNCcnt, no VGPR data)
      unsigned long long ga =
          (unsigned long long)(size_t)(W + (size_t)(k0 + skk) * Nout + n0 + snn);
      asm volatile("global_load_async_to_lds_b64 %0, %1, off"
                   :: "v"(lds_off), "v"(ga) : "memory");
    }
    // A fragment while the async engine fills LDS.
    // layout: elem j<8 -> K=j+8*half ; j>=8 -> K=16+(j-8)+8*half
    v16h a, b;
    if (active) {
      const float* Arow = A + (size_t)(m0 + l16) * K + k0;
      const float4* p = reinterpret_cast<const float4*>(Arow + 8 * half);
      float4 a0 = p[0], a1 = p[1];
      const float4* q = reinterpret_cast<const float4*>(Arow + 16 + 8 * half);
      float4 a2 = q[0], a3 = q[1];
      a[0]=(_Float16)a0.x;  a[1]=(_Float16)a0.y;  a[2]=(_Float16)a0.z;  a[3]=(_Float16)a0.w;
      a[4]=(_Float16)a1.x;  a[5]=(_Float16)a1.y;  a[6]=(_Float16)a1.z;  a[7]=(_Float16)a1.w;
      a[8]=(_Float16)a2.x;  a[9]=(_Float16)a2.y;  a[10]=(_Float16)a2.z; a[11]=(_Float16)a2.w;
      a[12]=(_Float16)a3.x; a[13]=(_Float16)a3.y; a[14]=(_Float16)a3.z; a[15]=(_Float16)a3.w;
    }
    asm volatile("s_wait_asynccnt 0" ::: "memory");  // this wave's copies landed in LDS
    __syncthreads();                                 // all waves' copies visible
    if (active) {
      // B fragment: elem j<8 -> K=j+8*half ; j>=8 -> K=16+(j-8)+8*half, N=l16
      int koff = 8 * half;
#pragma unroll
      for (int j = 0; j < 16; ++j) b[j] = (_Float16)Bsh[(koff + (j < 8 ? j : j + 8)) * BROW + l16];
      acc = __builtin_amdgcn_wmma_f32_16x16x32_f16(false, a, false, b, (short)0,
                                                   acc, false, false);
    }
  }
  if (!active) return;
  int col = n0 + l16;
  float bv = biasZ ? biasZ[col] : 0.f;
  float scale = 0.f, shift = 0.f;
  if (MODE == ACT_RELU_BN || MODE == ACT_RELU_BN_LEAKY || MODE == ACT_SILU_BN) {
    scale = ggZ[col] * rsqrtf(1.0f + EPSF);
    shift = bbZ[col];
  }
#pragma unroll
  for (int r = 0; r < 8; ++r) {
    int row = m0 + r + 8 * half;   // C/D layout: VGPR r -> M=r+8*half, N=lane%16
    float v = acc[r] + bv;
    if (MODE == ACT_RELU_BN || MODE == ACT_RELU_BN_LEAKY) {
      v = fmaxf(v, 0.f) * scale + shift;
      if (MODE == ACT_RELU_BN_LEAKY) v = leaky(v);
    } else if (MODE == ACT_TANH) {
      v = tanhf(v);
    } else if (MODE == ACT_SILU_BN) {
      v = (v / (1.f + __expf(-v))) * scale + shift;
    }
    out[(size_t)row * Nout + col] = v;
  }
}

// ---------- GATv2 edge kernels ----------
__global__ void gat_logits(const int* __restrict__ ei, const float* __restrict__ xl,
                           const float* __restrict__ xr, const float* __restrict__ att,
                           float* __restrict__ logit, unsigned* __restrict__ maxb) {
  int id = blockIdx.x * blockDim.x + threadIdx.x;
  if (id >= EPRIME * HEADS) return;
  int e = id >> 3, hd = id & 7;
  int s, d;
  if (e < N_EDGES) { s = ei[e]; d = ei[N_EDGES + e]; } else { s = d = e - N_EDGES; }
  const float* xls = xl + (size_t)s * HID + hd * CH;
  const float* xrd = xr + (size_t)d * HID + hd * CH;
  const float* at  = att + hd * CH;
  float acc = 0.f;
#pragma unroll 8
  for (int c = 0; c < CH; ++c) acc += leaky(xls[c] + xrd[c]) * at[c];
  logit[id] = acc;
  atomicMax(&maxb[d * HEADS + hd], fkey(acc));
}

__global__ void gat_expsum(const int* __restrict__ ei, const float* __restrict__ logit,
                           const unsigned* __restrict__ maxb, const float* __restrict__ xl,
                           float* __restrict__ den, float* __restrict__ num) {
  int id = blockIdx.x * blockDim.x + threadIdx.x;
  if (id >= EPRIME * HEADS) return;
  int e = id >> 3, hd = id & 7;
  int s, d;
  if (e < N_EDGES) { s = ei[e]; d = ei[N_EDGES + e]; } else { s = d = e - N_EDGES; }
  float m = funkey(maxb[d * HEADS + hd]);
  float w = __expf(logit[id] - m);
  atomicAdd(&den[d * HEADS + hd], w);
  const float* xls = xl + (size_t)s * HID + hd * CH;
  float* nd = num + (size_t)d * HID + hd * CH;
#pragma unroll 8
  for (int c = 0; c < CH; ++c) atomicAdd(&nd[c], w * xls[c]);
}

__global__ void gat_finalize(const float* __restrict__ num, const float* __restrict__ den,
                             const float* __restrict__ bias, float* __restrict__ h) {
  int id = blockIdx.x * blockDim.x + threadIdx.x;
  if (id >= N_NODES * HID) return;
  int n = id >> 8, f = id & 255;
  h[id] = num[id] / (den[n * HEADS + (f >> 5)] + 1e-16f) + bias[f];
}

// ---------- GIN ----------
__global__ void gin_agg(const int* __restrict__ ei, const float* __restrict__ h,
                        float* __restrict__ agg) {
  long id = (long)blockIdx.x * blockDim.x + threadIdx.x;
  if (id >= (long)N_EDGES * HID) return;
  int e = (int)(id >> 8), f = (int)(id & 255);
  int s = ei[e], d = ei[N_EDGES + e];
  atomicAdd(&agg[(size_t)d * HID + f], h[(size_t)s * HID + f]);
}

__global__ void add_eltwise(const float* __restrict__ a, const float* __restrict__ b,
                            float* __restrict__ c) {
  int id = blockIdx.x * blockDim.x + threadIdx.x;
  if (id < N_NODES * HID) c[id] = a[id] + b[id];
}

// ---------- global LayerNorm ----------
__global__ void reduce_sumsq(const float* __restrict__ h, int n, float* __restrict__ S) {
  float s = 0.f, q = 0.f;
  for (int i = blockIdx.x * blockDim.x + threadIdx.x; i < n; i += gridDim.x * blockDim.x) {
    float v = h[i]; s += v; q += v * v;
  }
  for (int off = 16; off; off >>= 1) {
    s += __shfl_down(s, off, 32);
    q += __shfl_down(q, off, 32);
  }
  if ((threadIdx.x & 31) == 0) { atomicAdd(&S[0], s); atomicAdd(&S[1], q); }
}

__global__ void ln_res_leaky(const float* __restrict__ hg, const float* __restrict__ res,
                             const float* __restrict__ S, const float* __restrict__ g,
                             const float* __restrict__ b, float* __restrict__ hout) {
  int id = blockIdx.x * blockDim.x + threadIdx.x;
  if (id >= N_NODES * HID) return;
  const float inv = 1.0f / ((float)N_NODES * (float)HID);
  float mu  = S[0] * inv;
  float var = fmaxf(S[1] * inv - mu * mu, 0.f);
  float sd  = sqrtf(var);
  int f = id & 255;
  float ln = (hg[id] - mu) / (sd + EPSF) * g[f] + b[f];
  hout[id] = leaky(ln + res[id]);
}

// ---------- pooling ----------
__global__ void gate_dot(const float* __restrict__ t, const float* __restrict__ W2,
                         const float* __restrict__ b2, const int* __restrict__ batch,
                         float* __restrict__ gate, unsigned* __restrict__ gmax) {
  int wid  = (blockIdx.x * blockDim.x + threadIdx.x) >> 5;
  int lane = threadIdx.x & 31;
  if (wid >= N_NODES) return;
  const float* row = t + (size_t)wid * HID;
  float s = 0.f;
  for (int c = lane; c < HID; c += 32) s += row[c] * W2[c];
  for (int off = 16; off; off >>= 1) s += __shfl_down(s, off, 32);
  if (lane == 0) {
    float gv = s + b2[0];
    gate[wid] = gv;
    atomicMax(&gmax[batch[wid]], fkey(gv));
  }
}

__global__ void pool_expsum(const float* __restrict__ gate, const unsigned* __restrict__ gmax,
                            float* __restrict__ gden, const float* __restrict__ h,
                            const int* __restrict__ batch, float* __restrict__ embnum) {
  int id = blockIdx.x * blockDim.x + threadIdx.x;
  if (id >= N_NODES * HID) return;
  int n = id >> 8, f = id & 255;
  int b = batch[n];
  float e = __expf(gate[n] - funkey(gmax[b]));
  if (f == 0) atomicAdd(&gden[b], e);
  atomicAdd(&embnum[b * HID + f], e * h[id]);
}

__global__ void emb_div(const float* __restrict__ embnum, const float* __restrict__ gden,
                        float* __restrict__ emb) {
  int id = blockIdx.x * blockDim.x + threadIdx.x;
  if (id >= NGRAPH * HID) return;
  emb[id] = embnum[id] / (gden[id >> 8] + 1e-16f);
}

// ---------- label head output ----------
__global__ void head_out(const float* __restrict__ zbuf, const float* __restrict__ W2,
                         const float* __restrict__ b2, float* __restrict__ out) {
  int wid  = (blockIdx.x * blockDim.x + threadIdx.x) >> 5;
  int lane = threadIdx.x & 31;
  if (wid >= NGRAPH * NLAB) return;
  int g = wid >> 6, o = wid & 63;
  const float* zr = zbuf + ((size_t)o * NGRAPH + g) * HID;
  const float* wr = W2 + (size_t)o * HID;
  float s = 0.f;
  for (int c = lane; c < HID; c += 32) s += zr[c] * wr[c];
  for (int off = 16; off; off >>= 1) s += __shfl_down(s, off, 32);
  if (lane == 0) out[g * NLAB + o] = s + b2[o];
}

// ---------- launch ----------
extern "C" void kernel_launch(void* const* d_in, const int* in_sizes, int n_in,
                              void* d_out, int out_size, void* d_ws, size_t ws_size,
                              hipStream_t stream) {
  (void)in_sizes; (void)n_in; (void)out_size; (void)ws_size;

  const float* x        = (const float*)d_in[0];
  const int*   ei       = (const int*)d_in[1];
  const int*   batch    = (const int*)d_in[2];
  const float* fp_W     = (const float*)d_in[3];
  const float* fp_b     = (const float*)d_in[4];
  const float* fp_g     = (const float*)d_in[5];
  const float* fp_be    = (const float*)d_in[6];
  const float* gat_Wl   = (const float*)d_in[7];
  const float* gat_bl   = (const float*)d_in[8];
  const float* gat_Wr   = (const float*)d_in[9];
  const float* gat_br   = (const float*)d_in[10];
  const float* gat_att  = (const float*)d_in[11];
  const float* gat_bias = (const float*)d_in[12];
  const float* gin_W    = (const float*)d_in[13];
  const float* gin_b    = (const float*)d_in[14];
  const float* gin_g    = (const float*)d_in[15];
  const float* gin_be   = (const float*)d_in[16];
  const float* ln_g     = (const float*)d_in[17];
  const float* ln_b     = (const float*)d_in[18];
  const float* res_W    = (const float*)d_in[19];
  const float* res_b    = (const float*)d_in[20];
  const float* pool_W1  = (const float*)d_in[21];
  const float* pool_b1  = (const float*)d_in[22];
  const float* pool_W2  = (const float*)d_in[23];
  const float* pool_b2  = (const float*)d_in[24];
  const float* head_W1  = (const float*)d_in[25];
  const float* head_b1  = (const float*)d_in[26];
  const float* head_g   = (const float*)d_in[27];
  const float* head_be  = (const float*)d_in[28];
  const float* head_W2  = (const float*)d_in[29];
  const float* head_b2  = (const float*)d_in[30];

  // workspace carve (all fp32)
  const size_t NH = (size_t)N_NODES * HID;
  float* Ah = (float*)d_ws;                 // current node features h
  float* Bh = Ah + NH;                      // temp (xl / hsum / res / t)
  float* Ch = Bh + NH;                      // temp (xr / agg / h_gin)
  float* Dh = Ch + NH;                      // temp (GAT numerator)
  float* logit  = Dh + NH;                              // EPRIME*HEADS
  unsigned* maxb = (unsigned*)(logit + (size_t)EPRIME * HEADS);  // N*HEADS
  float* den    = (float*)(maxb + (size_t)N_NODES * HEADS);      // N*HEADS
  float* zbuf   = den + (size_t)N_NODES * HEADS;        // NLAB*NGRAPH*HID
  float* embnum = zbuf + (size_t)NLAB * NGRAPH * HID;   // NGRAPH*HID
  float* emb    = embnum + NGRAPH * HID;                // NGRAPH*HID
  float* gate   = emb + NGRAPH * HID;                   // N
  float* S      = gate + N_NODES;                       // 2
  unsigned* gmax = (unsigned*)(S + 2);                  // NGRAPH
  float* gden   = (float*)(gmax + NGRAPH);              // NGRAPH

  auto cdiv = [](long a, long b) { return (int)((a + b - 1) / b); };
  const int NHi = N_NODES * HID;
  const int EH  = EPRIME * HEADS;

  dim3 gridBig(HID / 16, (N_NODES / 16 + 7) / 8, 1);  // 16 x 125

  // feature_proj: h = BN(relu(x @ fp_W + fp_b))
  wmma_gemm<ACT_RELU_BN><<<gridBig, 256, 0, stream>>>(
      x, fp_W, fp_b, fp_g, fp_be, Ah, N_NODES, DIN, HID, 0, 0, 0, 0, 0);

  for (int i = 0; i < 2; ++i) {
    const float* Wl  = gat_Wl + (size_t)i * HID * HID;
    const float* bl  = gat_bl + (size_t)i * HID;
    const float* Wr  = gat_Wr + (size_t)i * HID * HID;
    const float* br  = gat_br + (size_t)i * HID;
    const float* att = gat_att + (size_t)i * HEADS * CH;
    const float* gb  = gat_bias + (size_t)i * HID;
    const float* gW  = gin_W + (size_t)i * HID * HID;
    const float* gbi = gin_b + (size_t)i * HID;
    const float* ggi = gin_g + (size_t)i * HID;
    const float* gbe = gin_be + (size_t)i * HID;
    const float* lg  = ln_g + (size_t)i * HID;
    const float* lb  = ln_b + (size_t)i * HID;
    const float* rW  = res_W + (size_t)i * HID * HID;
    const float* rb  = res_b + (size_t)i * HID;

    // xl = h@Wl+bl, xr = h@Wr+br
    wmma_gemm<ACT_NONE><<<gridBig, 256, 0, stream>>>(Ah, Wl, bl, nullptr, nullptr, Bh,
                                                     N_NODES, HID, HID, 0, 0, 0, 0, 0);
    wmma_gemm<ACT_NONE><<<gridBig, 256, 0, stream>>>(Ah, Wr, br, nullptr, nullptr, Ch,
                                                     N_NODES, HID, HID, 0, 0, 0, 0, 0);
    // GATv2 attention aggregation
    fill_key<<<cdiv(N_NODES * HEADS, 256), 256, 0, stream>>>(maxb, -3.0e38f, N_NODES * HEADS);
    fill_f32<<<cdiv(N_NODES * HEADS, 256), 256, 0, stream>>>(den, 0.f, N_NODES * HEADS);
    fill_f32<<<cdiv(NHi, 256), 256, 0, stream>>>(Dh, 0.f, NHi);
    gat_logits<<<cdiv(EH, 256), 256, 0, stream>>>(ei, Bh, Ch, att, logit, maxb);
    gat_expsum<<<cdiv(EH, 256), 256, 0, stream>>>(ei, logit, maxb, Bh, den, Dh);
    gat_finalize<<<cdiv(NHi, 256), 256, 0, stream>>>(Dh, den, gb, Ah);
    // GIN: agg over original edges, hsum = h + agg, nn(hsum), leaky
    fill_f32<<<cdiv(NHi, 256), 256, 0, stream>>>(Ch, 0.f, NHi);
    gin_agg<<<cdiv((long)N_EDGES * HID, 256), 256, 0, stream>>>(ei, Ah, Ch);
    add_eltwise<<<cdiv(NHi, 256), 256, 0, stream>>>(Ah, Ch, Bh);
    wmma_gemm<ACT_RELU_BN_LEAKY><<<gridBig, 256, 0, stream>>>(
        Bh, gW, gbi, ggi, gbe, Ch, N_NODES, HID, HID, 0, 0, 0, 0, 0);
    // global LayerNorm + residual linear + leaky
    fill_f32<<<1, 256, 0, stream>>>(S, 0.f, 2);
    reduce_sumsq<<<1024, 256, 0, stream>>>(Ch, NHi, S);
    wmma_gemm<ACT_NONE><<<gridBig, 256, 0, stream>>>(Ch, rW, rb, nullptr, nullptr, Bh,
                                                     N_NODES, HID, HID, 0, 0, 0, 0, 0);
    ln_res_leaky<<<cdiv(NHi, 256), 256, 0, stream>>>(Ch, Bh, S, lg, lb, Ah);
  }

  // GlobalAttention pooling
  wmma_gemm<ACT_TANH><<<gridBig, 256, 0, stream>>>(Ah, pool_W1, pool_b1, nullptr, nullptr,
                                                   Bh, N_NODES, HID, HID, 0, 0, 0, 0, 0);
  fill_key<<<1, 256, 0, stream>>>(gmax, -3.0e38f, NGRAPH);
  fill_f32<<<1, 256, 0, stream>>>(gden, 0.f, NGRAPH);
  gate_dot<<<cdiv((long)N_NODES * 32, 256), 256, 0, stream>>>(Bh, pool_W2, pool_b2, batch,
                                                              gate, gmax);
  fill_f32<<<cdiv(NGRAPH * HID, 256), 256, 0, stream>>>(embnum, 0.f, NGRAPH * HID);
  pool_expsum<<<cdiv(NHi, 256), 256, 0, stream>>>(gate, gmax, gden, Ah, batch, embnum);
  emb_div<<<cdiv(NGRAPH * HID, 256), 256, 0, stream>>>(embnum, gden, emb);

  // label heads: z_o = BN(silu(emb @ head_W1[o] + head_b1[o])), batched over o
  dim3 gridH(HID / 16, 1, NLAB);
  wmma_gemm<ACT_SILU_BN><<<gridH, 256, 0, stream>>>(
      emb, head_W1, head_b1, head_g, head_be, zbuf, NGRAPH, HID, HID,
      0, (long)HID * HID, HID, HID, (long)NGRAPH * HID);
  head_out<<<cdiv((long)NGRAPH * NLAB * 32, 256), 256, 0, stream>>>(
      zbuf, head_W2, head_b2, (float*)d_out);
}
